// ERCGNN_54408645706106
// MI455X (gfx1250) — matrix operand order repaired and checked by reading, not compiled, not run on hardware
//
#include <hip/hip_runtime.h>

#define DD 256
#define HH 8
#define DHH 32
#define NCLS 7
#define BN_EPS 1e-9f

typedef __attribute__((ext_vector_type(16))) __bf16       v16bf;
typedef __attribute__((ext_vector_type(8)))  float        v8f;
typedef __attribute__((ext_vector_type(4)))  unsigned int u32x4;
typedef __attribute__((ext_vector_type(4)))  float        f32x4;

static __device__ __forceinline__ unsigned short f32_to_bf16(float f) {
    unsigned int u = __float_as_uint(f);
    unsigned int r = u + 0x7FFFu + ((u >> 16) & 1u);   // round-to-nearest-even
    return (unsigned short)(r >> 16);
}

union FragBF { u32x4 u[2]; v16bf v; };

// ---------------------------------------------------------------------------
// WMMA GEMM: C[M,256] = relu?( bf16(A)[M,256] @ Wt^T + bias )
// A: bf16 row-major [M,256]; Wt: bf16 [256,256] where row n holds W[:,n].
// Block = 128 threads (4 waves); block covers 32 rows, wave covers 64 cols.
// Each wave: 2 row-tiles x 4 col-tiles = 8 accumulators; every B fragment
// feeds two independent WMMAs so load latency overlaps compute.
// Tail block (M=50000 = 1562*32 + 16): WMMAs run with EXEC all-ones on
// garbage rows (loads stay inside the workspace), stores are row-guarded.
// ---------------------------------------------------------------------------
__global__ __launch_bounds__(128)
void k_gemm256(const unsigned short* __restrict__ A,
               const unsigned short* __restrict__ Wt,
               const float* __restrict__ bias,
               float* __restrict__ C,
               int M, int relu)
{
    const int lane  = threadIdx.x & 31;
    const int wave  = threadIdx.x >> 5;      // 0..3
    const int m0    = blockIdx.x << 5;       // 32 rows per block
    const int n0    = wave << 6;             // 64 cols per wave
    const int mrow  = lane & 15;
    const int khalf = lane >> 4;             // 0 or 1

    v8f acc[2][4] = {};

    const unsigned short* arow0 = A + (size_t)(m0 + mrow) * DD;
    const unsigned short* arow1 = arow0 + (size_t)16 * DD;
    const unsigned short* wbase = Wt + (size_t)(n0 + mrow) * DD;

#pragma unroll
    for (int kb = 0; kb < DD; kb += 32) {
        // A frags: lane holds K = {kb+8*khalf..+7} and {kb+16+8*khalf..+7}
        FragBF a0, a1;
        a0.u[0] = *(const u32x4*)(arow0 + kb + khalf * 8);
        a0.u[1] = *(const u32x4*)(arow0 + kb + 16 + khalf * 8);
        a1.u[0] = *(const u32x4*)(arow1 + kb + khalf * 8);
        a1.u[1] = *(const u32x4*)(arow1 + kb + 16 + khalf * 8);
        // B frags: lane holds K = {kb+8*khalf..+7} and {kb+16+8*khalf..+7}
        // Load all four col-tiles first, then issue all 8 WMMAs.
        const int bo = kb + khalf * 8;
        FragBF b[4];
#pragma unroll
        for (int t = 0; t < 4; ++t) {
            const unsigned short* wr = wbase + (size_t)(t * 16) * DD;
            b[t].u[0] = *(const u32x4*)(wr + bo);
            b[t].u[1] = *(const u32x4*)(wr + bo + 16);
        }
#pragma unroll
        for (int t = 0; t < 4; ++t) {
            acc[0][t] = __builtin_amdgcn_wmma_f32_16x16x32_bf16(
                false, a0.v, false, b[t].v, (short)0, acc[0][t], false, false);
            acc[1][t] = __builtin_amdgcn_wmma_f32_16x16x32_bf16(
                false, a1.v, false, b[t].v, (short)0, acc[1][t], false, false);
        }
    }

#pragma unroll
    for (int t = 0; t < 4; ++t) {
        const int n = n0 + t * 16 + mrow;
        const float bv = bias[n];
#pragma unroll
        for (int v = 0; v < 8; ++v) {
            const int m = m0 + v + 8 * khalf;   // C/D layout: row = v + 8*(lane>>4)
            float v0 = acc[0][t][v] + bv;
            float v1 = acc[1][t][v] + bv;
            if (relu) { v0 = fmaxf(v0, 0.0f); v1 = fmaxf(v1, 0.0f); }
            if (m < M)      C[(size_t)m * DD + n] = v0;
            if (m + 16 < M) C[(size_t)(m + 16) * DD + n] = v1;
        }
    }
}

// ---------------------------------------------------------------------------
// Conversions / weight packing
// ---------------------------------------------------------------------------
__global__ void k_tobf16(const float* __restrict__ x, unsigned short* __restrict__ y, int n) {
    int i = blockIdx.x * 256 + threadIdx.x;
    if (i < n) y[i] = f32_to_bf16(x[i]);
}

// GAT weight slice [H, D, DH] -> wt[oc=h*32+k][d]  (bf16, transposed)
__global__ void k_wt_gat(const float* __restrict__ Ws, unsigned short* __restrict__ wt) {
    int idx = blockIdx.x * 256 + threadIdx.x;        // 65536 total
    int d = idx & 255, oc = idx >> 8;
    int h = oc >> 5, k = oc & 31;
    wt[(size_t)oc * DD + d] = f32_to_bf16(Ws[(size_t)h * (DD * DHH) + (size_t)d * DHH + k]);
}

// Generic [K=256, Dout=256] -> wt[n][k] (bf16, transposed)
__global__ void k_wt_t(const float* __restrict__ W, unsigned short* __restrict__ wt) {
    int idx = blockIdx.x * 256 + threadIdx.x;
    int k = idx & 255, n = idx >> 8;
    wt[(size_t)n * DD + k] = f32_to_bf16(W[(size_t)k * DD + n]);
}

// ---------------------------------------------------------------------------
// Attention: att_s/att_n both from f_self (faithful to reference), lrelu 0.2
// ---------------------------------------------------------------------------
__global__ void k_att(const float* __restrict__ fs,
                      const float* __restrict__ a_s, const float* __restrict__ a_n,
                      float* __restrict__ atts, float* __restrict__ attn, int Nn) {
    int idx = blockIdx.x * 256 + threadIdx.x;
    if (idx >= Nn * HH) return;
    int n = idx >> 3, h = idx & 7;
    const float* p   = fs + (size_t)n * DD + h * DHH;
    const float* as_ = a_s + h * DHH;
    const float* an_ = a_n + h * DHH;
    float ss = 0.f, sn = 0.f;
#pragma unroll
    for (int k = 0; k < DHH; ++k) { float v = p[k]; ss += v * as_[k]; sn += v * an_[k]; }
    atts[idx] = ss > 0.f ? ss : 0.2f * ss;
    attn[idx] = sn > 0.f ? sn : 0.2f * sn;
}

// ---------------------------------------------------------------------------
// Edge scatter kernels: 64 lanes per edge, float4 per lane, atomic f32 adds.
// Feature matrix (51 MB) is L2-resident on MI455X (192 MB L2).
// ---------------------------------------------------------------------------
__global__ void k_gat_edge(const int* __restrict__ row, const int* __restrict__ col,
                           const float* __restrict__ vals,
                           const float* __restrict__ atts, const float* __restrict__ attn,
                           const float* __restrict__ fneigh, float* __restrict__ agg, int Ee) {
    int e = blockIdx.x * 4 + (threadIdx.x >> 6);
    if (e >= Ee) return;
    int j = threadIdx.x & 63;
    int r = row[e], c = col[e];
    int h = j >> 3;                              // (j*4)/32
    float coef = (atts[(size_t)r * HH + h] + attn[(size_t)c * HH + h]) * vals[e];
    f32x4 fn = *(const f32x4*)(fneigh + (size_t)c * DD + j * 4);
    float* dst = agg + (size_t)r * DD + j * 4;
    atomicAdd(dst + 0, coef * fn.x);
    atomicAdd(dst + 1, coef * fn.y);
    atomicAdd(dst + 2, coef * fn.z);
    atomicAdd(dst + 3, coef * fn.w);
}

__global__ void k_gcn_edge(const int* __restrict__ row, const int* __restrict__ col,
                           const float* __restrict__ vals,
                           const float* __restrict__ hmat, float* __restrict__ agg, int Ee) {
    int e = blockIdx.x * 4 + (threadIdx.x >> 6);
    if (e >= Ee) return;
    int j = threadIdx.x & 63;
    int r = row[e], c = col[e];
    float coef = vals[e];
    f32x4 hv = *(const f32x4*)(hmat + (size_t)c * DD + j * 4);
    float* dst = agg + (size_t)r * DD + j * 4;
    atomicAdd(dst + 0, coef * hv.x);
    atomicAdd(dst + 1, coef * hv.y);
    atomicAdd(dst + 2, coef * hv.z);
    atomicAdd(dst + 3, coef * hv.w);
}

// ---------------------------------------------------------------------------
// BatchNorm (training-mode, biased var). stats[0:256]=sum, stats[256:512]=sumsq
// ---------------------------------------------------------------------------
__global__ void k_bn_stats(const float* __restrict__ x, float* __restrict__ stats,
                           int M, int relu) {
    int c = threadIdx.x;
    int r0 = blockIdx.x * 64;
    int rend = min(r0 + 64, M);
    float s = 0.f, s2 = 0.f;
    for (int r = r0; r < rend; ++r) {
        float v = x[(size_t)r * DD + c];
        if (relu) v = fmaxf(v, 0.f);
        s += v; s2 += v * v;
    }
    atomicAdd(&stats[c], s);
    atomicAdd(&stats[DD + c], s2);
}

// Fused: writes f32 result (for classifier) AND bf16 copy (next layer's GEMM A)
__global__ void k_bn_norm(const float* __restrict__ x, const float* __restrict__ stats,
                          float* __restrict__ y, unsigned short* __restrict__ ybf,
                          int M, int relu) {
    size_t idx = (size_t)blockIdx.x * 256 + threadIdx.x;
    if (idx >= (size_t)M * DD) return;
    int c = (int)(idx & 255);
    float inv = 1.0f / (float)M;
    float mu  = stats[c] * inv;
    float var = stats[DD + c] * inv - mu * mu;
    float v = x[idx];
    if (relu) v = fmaxf(v, 0.f);
    float r = (v - mu) * rsqrtf(var + BN_EPS);
    y[idx] = r;
    ybf[idx] = f32_to_bf16(r);
}

// ---------------------------------------------------------------------------
// Classifier: out[n,:7] initialized to bias, then accumulated per branch.
// ---------------------------------------------------------------------------
__global__ void k_out_init(float* __restrict__ out, const float* __restrict__ b, int n) {
    int i = blockIdx.x * 256 + threadIdx.x;
    if (i < n) out[i] = b[i % NCLS];
}

__global__ void k_cls_accum(const float* __restrict__ fb, const float* __restrict__ W,
                            int coloff, float* __restrict__ out) {
    __shared__ float sred[256];
    int n = blockIdx.x;
    int t = threadIdx.x;
    float v = fb[(size_t)n * DD + t];
    const float* wrow = W + (size_t)(coloff + t) * NCLS;
    float acc[NCLS];
#pragma unroll
    for (int c = 0; c < NCLS; ++c) acc[c] = v * wrow[c];
#pragma unroll
    for (int c = 0; c < NCLS; ++c) {
        sred[t] = acc[c];
        __syncthreads();
        for (int s = 128; s > 0; s >>= 1) {
            if (t < s) sred[t] += sred[t + s];
            __syncthreads();
        }
        if (t == 0) out[(size_t)n * NCLS + c] += sred[0];
        __syncthreads();
    }
}

// ---------------------------------------------------------------------------
extern "C" void kernel_launch(void* const* d_in, const int* in_sizes, int n_in,
                              void* d_out, int out_size, void* d_ws, size_t ws_size,
                              hipStream_t stream) {
    (void)n_in; (void)out_size; (void)ws_size;
    const float* f_in   = (const float*)d_in[0];
    const int*   erow   = (const int*)d_in[1];
    const int*   ecol   = (const int*)d_in[2];
    const float* evals  = (const float*)d_in[3];
    const float* gat_Ws = (const float*)d_in[4];
    const float* gat_bs = (const float*)d_in[5];
    const float* gat_Wn = (const float*)d_in[6];
    const float* gat_bn = (const float*)d_in[7];
    const float* gat_as = (const float*)d_in[8];
    const float* gat_an = (const float*)d_in[9];
    const float* gcn_W  = (const float*)d_in[10];
    const float* gcn_b  = (const float*)d_in[11];
    const float* self_W = (const float*)d_in[12];
    const float* self_b = (const float*)d_in[13];
    const float* cls_W  = (const float*)d_in[14];
    const float* cls_b  = (const float*)d_in[15];
    float* out = (float*)d_out;

    const int Nn = in_sizes[0] / DD;     // 50000
    const int Ee = in_sizes[1] / 4;      // 400000
    const size_t FB = (size_t)Nn * DD * sizeof(float);

    // workspace carve (all 16B-aligned)
    char* w = (char*)d_ws;
    float*          fbuf  = (float*)(w);                                   // FB
    unsigned short* fbf   = (unsigned short*)(w + FB);                     // FB/2
    float*          g1    = (float*)(w + FB + FB / 2);                     // FB
    float*          g2    = (float*)(w + 2 * FB + FB / 2);                 // FB
    float*          agg   = (float*)(w + 3 * FB + FB / 2);                 // FB
    float*          atts  = (float*)(w + 4 * FB + FB / 2);
    float*          attn  = (float*)(w + 4 * FB + FB / 2 + (size_t)Nn * HH * 4);
    float*          stats = (float*)(w + 4 * FB + FB / 2 + 2 * (size_t)Nn * HH * 4);
    unsigned short* wt    = (unsigned short*)(w + 4 * FB + FB / 2 + 2 * (size_t)Nn * HH * 4 + 4096);
    unsigned short* finbf = (unsigned short*)(w + 4 * FB + FB / 2 + 2 * (size_t)Nn * HH * 4 + 4096
                                                + (size_t)DD * DD * 2);    // FB/2

    const int gemmGrid  = (Nn + 31) / 32;             // 1563 (tail-guarded)
    const int elemGrid  = (Nn * DD + 255) / 256;      // 50000
    const int wtGrid    = (DD * DD) / 256;            // 256
    const int attGrid   = (Nn * HH + 255) / 256;
    const int edgeGrid  = (Ee + 3) / 4;
    const int statsGrid = (Nn + 63) / 64;

    k_out_init<<<(Nn * NCLS + 255) / 256, 256, 0, stream>>>(out, cls_b, Nn * NCLS);

    // bf16(f_in): layer-0 GEMM input of all five branches, converted once.
    k_tobf16<<<elemGrid, 256, 0, stream>>>(f_in, finbf, Nn * DD);

    // ---- GAT branch (adj 0) ----
    for (int l = 0; l < 2; ++l) {
        const unsigned short* Abf = (l == 0) ? finbf : fbf;
        k_wt_gat<<<wtGrid, 256, 0, stream>>>(gat_Ws + (size_t)l * HH * DD * DHH, wt);
        k_gemm256<<<gemmGrid, 128, 0, stream>>>(Abf, wt, gat_bs + l * DD, g1, Nn, 1);  // f_self
        k_wt_gat<<<wtGrid, 256, 0, stream>>>(gat_Wn + (size_t)l * HH * DD * DHH, wt);
        k_gemm256<<<gemmGrid, 128, 0, stream>>>(Abf, wt, gat_bn + l * DD, g2, Nn, 1);  // f_neigh
        k_att<<<attGrid, 256, 0, stream>>>(g1, gat_as + l * DD, gat_an + l * DD, atts, attn, Nn);
        hipMemsetAsync(agg, 0, FB, stream);
        k_gat_edge<<<edgeGrid, 256, 0, stream>>>(erow, ecol, evals, atts, attn, g2, agg, Ee);
        hipMemsetAsync(stats, 0, 2 * DD * sizeof(float), stream);
        k_bn_stats<<<statsGrid, 256, 0, stream>>>(agg, stats, Nn, 0);
        k_bn_norm<<<elemGrid, 256, 0, stream>>>(agg, stats, fbuf, fbf, Nn, 0);
    }
    k_cls_accum<<<Nn, 256, 0, stream>>>(fbuf, cls_W, 0, out);

    // ---- 3 vanilla GCN branches (adjs 1..3) ----
    for (int g = 0; g < 3; ++g) {
        for (int l = 0; l < 2; ++l) {
            const unsigned short* Abf = (l == 0) ? finbf : fbf;
            k_wt_t<<<wtGrid, 256, 0, stream>>>(gcn_W + (size_t)(g * 2 + l) * DD * DD, wt);
            k_gemm256<<<gemmGrid, 128, 0, stream>>>(Abf, wt, gcn_b + (g * 2 + l) * DD, g1, Nn, 0);
            hipMemsetAsync(agg, 0, FB, stream);
            k_gcn_edge<<<edgeGrid, 256, 0, stream>>>(erow + (size_t)(g + 1) * Ee,
                                                     ecol + (size_t)(g + 1) * Ee,
                                                     evals + (size_t)(g + 1) * Ee, g1, agg, Ee);
            hipMemsetAsync(stats, 0, 2 * DD * sizeof(float), stream);
            k_bn_stats<<<statsGrid, 256, 0, stream>>>(agg, stats, Nn, 1);   // BN(relu(agg))
            k_bn_norm<<<elemGrid, 256, 0, stream>>>(agg, stats, fbuf, fbf, Nn, 1);
        }
        k_cls_accum<<<Nn, 256, 0, stream>>>(fbuf, cls_W, (1 + g) * DD, out);
    }

    // ---- self perceptron branch ----
    k_wt_t<<<wtGrid, 256, 0, stream>>>(self_W, wt);
    k_gemm256<<<gemmGrid, 128, 0, stream>>>(finbf, wt, self_b, g1, Nn, 1);  // relu in GEMM
    hipMemsetAsync(stats, 0, 2 * DD * sizeof(float), stream);
    k_bn_stats<<<statsGrid, 256, 0, stream>>>(g1, stats, Nn, 0);
    k_bn_norm<<<elemGrid, 256, 0, stream>>>(g1, stats, fbuf, fbf, Nn, 0);
    k_cls_accum<<<Nn, 256, 0, stream>>>(fbuf, cls_W, 4 * DD, out);
}
